// M_MGN_85581518340279
// MI455X (gfx1250) — matrix operand order repaired and checked by reading, not compiled, not run on hardware
//
#include <hip/hip_runtime.h>

// ---------------------------------------------------------------------------
// MI455X (gfx1250, wave32) fused implementation of
//   out = a + x @ (V^T V) + sum_l s_l * (tanh(Z_l) @ Wk_l),  Z_l = x Wk_l^T + bk_l
// All large GEMMs run on v_wmma_f32_16x16x32_bf16 (f32 accumulate).
// Shapes are compile-time constants so all fragment offsets fold into VMEM
// immediate offsets (one base address per operand panel in the K-loop).
// ---------------------------------------------------------------------------

constexpr int CB = 4096;  // batch
constexpr int CD = 1024;  // feature dim
constexpr int CH = 4096;  // hidden dim
constexpr int CL = 8;     // layers

typedef __attribute__((ext_vector_type(16))) __bf16 v16bf;
typedef __attribute__((ext_vector_type(8)))  __bf16 v8bf;
typedef __attribute__((ext_vector_type(8)))  float  v8f;

__device__ __forceinline__ v8f wmma_bf16(v16bf a, v16bf b, v8f c) {
  // (neg_a, A, neg_b, B, c_mod, C, reuse_a, reuse_b)
  return __builtin_amdgcn_wmma_f32_16x16x32_bf16(false, a, false, b,
                                                 (short)0, c, false, false);
}

__device__ __forceinline__ float logcosh_f(float z) {
  float az = fabsf(z);
  return az + log1pf(__expf(-2.0f * az)) - 0.6931471805599453f;
}

// ---- K-loop: 4x4 WMMA tiles, compile-time leading dimension ---------------
// A fragment 16x32 (MxK): lanes 0-15 row M=lane, K {0..7,16..23};
//                         lanes 16-31 row M=lane-16, K {8..15,24..31}.
// B fragment 32x16 (KxN) from B^T [N][K]: lanes 0-15 col N=lane, K {0..7,16..23};
//                                         lanes 16-31 col N=lane-16, K {8..15,24..31}.
template <int LDK, int K>
__device__ __forceinline__ void mgn_kloop(const __bf16* __restrict__ ap,
                                          const __bf16* __restrict__ bp,
                                          int lane, v8f (&acc)[4][4]) {
  const int h = lane >> 4, ml = lane & 15;
  const __bf16* a_lane = ap + (size_t)ml * LDK + h * 8;   // +8 elems for hi half
  const __bf16* b_lane = bp + (size_t)ml * LDK + h * 8;   // +8 elems for hi half
  for (int k0 = 0; k0 < K; k0 += 32) {
    __builtin_prefetch(a_lane + k0 + 64, 0, 3);
    __builtin_prefetch(b_lane + k0 + 64, 0, 3);
    v16bf af[4], bf[4];
#pragma unroll
    for (int t = 0; t < 4; t++) {
      const __bf16* p = a_lane + (size_t)(16 * t) * LDK + k0;
      v8bf lo = *reinterpret_cast<const v8bf*>(p);
      v8bf hi = *reinterpret_cast<const v8bf*>(p + 16);
#pragma unroll
      for (int i = 0; i < 8; i++) { af[t][i] = lo[i]; af[t][i + 8] = hi[i]; }
    }
#pragma unroll
    for (int t = 0; t < 4; t++) {
      const __bf16* p = b_lane + (size_t)(16 * t) * LDK + k0;
      v8bf lo = *reinterpret_cast<const v8bf*>(p);
      v8bf hi = *reinterpret_cast<const v8bf*>(p + 16);
#pragma unroll
      for (int i = 0; i < 8; i++) { bf[t][i] = lo[i]; bf[t][i + 8] = hi[i]; }
    }
#pragma unroll
    for (int i = 0; i < 4; i++)
#pragma unroll
      for (int j = 0; j < 4; j++) acc[i][j] = wmma_bf16(af[i], bf[j], acc[i][j]);
  }
}

// ---- Utility kernels ------------------------------------------------------
__global__ void mgn_zero_f32(float* __restrict__ p, int n) {
  int i = blockIdx.x * blockDim.x + threadIdx.x;
  if (i < n) p[i] = 0.0f;
}

// dst (bf16) = src (f32), 8 elements per thread, vectorized.
__global__ void mgn_cvt_bf16(const float* __restrict__ src, __bf16* __restrict__ dst,
                             int n8) {
  int i = blockIdx.x * blockDim.x + threadIdx.x;
  if (i >= n8) return;
  const float4* s = reinterpret_cast<const float4*>(src) + (size_t)2 * i;
  float4 u0 = s[0], u1 = s[1];
  v8bf o;
  o[0] = (__bf16)u0.x; o[1] = (__bf16)u0.y; o[2] = (__bf16)u0.z; o[3] = (__bf16)u0.w;
  o[4] = (__bf16)u1.x; o[5] = (__bf16)u1.y; o[6] = (__bf16)u1.z; o[7] = (__bf16)u1.w;
  reinterpret_cast<v8bf*>(dst)[i] = o;
}

// One pass over src[R][C] (f32): dstD[R][C] = bf16(src), dstT[C][R] = bf16(src^T).
// R, C multiples of 32. 256 threads; LDS 32x33 tile for conflict-free transpose.
__global__ void mgn_cvt_dual(const float* __restrict__ src,
                             __bf16* __restrict__ dstD, __bf16* __restrict__ dstT,
                             int R, int C) {
  __shared__ float tile[32][33];
  const int c0 = blockIdx.x * 32, r0 = blockIdx.y * 32;
  const int tx = threadIdx.x & 31, ty = threadIdx.x >> 5;  // ty in [0,8)
#pragma unroll
  for (int i = ty; i < 32; i += 8) {
    float v = src[(size_t)(r0 + i) * C + c0 + tx];
    tile[i][tx] = v;
    dstD[(size_t)(r0 + i) * C + c0 + tx] = (__bf16)v;
  }
  __syncthreads();
#pragma unroll
  for (int i = ty; i < 32; i += 8)
    dstT[(size_t)(c0 + i) * R + r0 + tx] = (__bf16)tile[tx][i];
}

// ---- GEMM kernels ---------------------------------------------------------
// Tiling: 256 threads = 8 waves (2x4); wave tile 64x64 = 4x4 WMMA tiles;
// block tile 128(M) x 256(N); K stepped by 32; bf16 operands, f32 accumulate.

#define MGN_GEMM_PROLOG()                                        \
  const int lane = threadIdx.x & 31, wid = threadIdx.x >> 5;     \
  const int m0 = blockIdx.y * 128 + (wid >> 2) * 64;             \
  const int n0 = blockIdx.x * 256 + (wid & 3) * 64;              \
  v8f acc[4][4] = {};

// Vsq[D][D] (bf16) = VbT @ VbT^T :  Vsq[d1,d2] = sum_h V[h,d1] V[h,d2]
__global__ __launch_bounds__(256) void mgn_gemm_vsq(const __bf16* __restrict__ VbT,
                                                    __bf16* __restrict__ Vsq) {
  MGN_GEMM_PROLOG();
  mgn_kloop<CH, CH>(VbT + (size_t)m0 * CH, VbT + (size_t)n0 * CH, lane, acc);
  const int half = lane >> 4, nl = lane & 15;
#pragma unroll
  for (int i = 0; i < 4; i++)
#pragma unroll
    for (int r = 0; r < 8; r++) {
      const int row = m0 + 16 * i + 8 * half + r;
#pragma unroll
      for (int j = 0; j < 4; j++)
        Vsq[(size_t)row * CD + n0 + 16 * j + nl] = (__bf16)acc[i][j][r];
    }
}

// out[B][D] = a[d] + xb @ Vsq  (Vsq symmetric => itself is its transposed-B)
__global__ __launch_bounds__(256) void mgn_gemm_out0(const __bf16* __restrict__ xb,
                                                     const __bf16* __restrict__ Vsq,
                                                     const float* __restrict__ avec,
                                                     float* __restrict__ out) {
  MGN_GEMM_PROLOG();
  mgn_kloop<CD, CD>(xb + (size_t)m0 * CD, Vsq + (size_t)n0 * CD, lane, acc);
  const int half = lane >> 4, nl = lane & 15;
#pragma unroll
  for (int i = 0; i < 4; i++)
#pragma unroll
    for (int r = 0; r < 8; r++) {
      const int row = m0 + 16 * i + 8 * half + r;
#pragma unroll
      for (int j = 0; j < 4; j++) {
        const int col = n0 + 16 * j + nl;
        out[(size_t)row * CD + col] = avec[col] + acc[i][j][r];
      }
    }
}

// Z = xb @ Wkb^T + bk_l ; sigma = tanh(Z) (bf16) ; sbuf[b] += sum_h logcosh(Z)
__global__ __launch_bounds__(256) void mgn_gemm1_act(const __bf16* __restrict__ xb,  // [B][D]
                                                     const __bf16* __restrict__ Wkb, // [H][D]
                                                     const float* __restrict__ bkl,  // [H]
                                                     __bf16* __restrict__ sigma,     // [B][H]
                                                     float* __restrict__ sbuf) {     // [B]
  MGN_GEMM_PROLOG();
  mgn_kloop<CD, CD>(xb + (size_t)m0 * CD, Wkb + (size_t)n0 * CD, lane, acc);
  const int half = lane >> 4, nl = lane & 15;
#pragma unroll
  for (int i = 0; i < 4; i++) {
#pragma unroll
    for (int r = 0; r < 8; r++) {
      const int row = m0 + 16 * i + 8 * half + r;
      float rs = 0.0f;
#pragma unroll
      for (int j = 0; j < 4; j++) {
        const int col = n0 + 16 * j + nl;
        const float z = acc[i][j][r] + bkl[col];
        rs += logcosh_f(z);
        sigma[(size_t)row * CH + col] = (__bf16)tanhf(z);
      }
      rs += __shfl_xor(rs, 1, 32);
      rs += __shfl_xor(rs, 2, 32);
      rs += __shfl_xor(rs, 4, 32);
      rs += __shfl_xor(rs, 8, 32);
      if (nl == 0) atomicAdd(&sbuf[row], rs);
    }
  }
}

// out[b][d] += sbuf[b] * (sigma @ Wk_l)[b][d]   (WkT = Wk_l^T, [D][H] bf16)
__global__ __launch_bounds__(256) void mgn_gemm2_acc(const __bf16* __restrict__ sigma, // [B][H]
                                                     const __bf16* __restrict__ WkT,   // [D][H]
                                                     const float* __restrict__ sbuf,   // [B]
                                                     float* __restrict__ out) {        // [B][D]
  MGN_GEMM_PROLOG();
  mgn_kloop<CH, CH>(sigma + (size_t)m0 * CH, WkT + (size_t)n0 * CH, lane, acc);
  const int half = lane >> 4, nl = lane & 15;
#pragma unroll
  for (int i = 0; i < 4; i++)
#pragma unroll
    for (int r = 0; r < 8; r++) {
      const int row = m0 + 16 * i + 8 * half + r;
      const float sv = sbuf[row];
#pragma unroll
      for (int j = 0; j < 4; j++) {
        const int col = n0 + 16 * j + nl;
        out[(size_t)row * CD + col] += sv * acc[i][j][r];
      }
    }
}

// ---------------------------------------------------------------------------
extern "C" void kernel_launch(void* const* d_in, const int* in_sizes, int n_in,
                              void* d_out, int out_size, void* d_ws, size_t ws_size,
                              hipStream_t stream) {
  (void)in_sizes; (void)n_in; (void)out_size; (void)ws_size;
  const int B = CB, D = CD, H = CH, L = CL;

  const float* x  = (const float*)d_in[0];
  const float* Wk = (const float*)d_in[1];  // [L][H][D]
  const float* bk = (const float*)d_in[2];  // [L][H]
  const float* V  = (const float*)d_in[3];  // [H][D]
  const float* a  = (const float*)d_in[4];  // [D]
  float* out = (float*)d_out;               // [B][D]

  // workspace layout (~66 MB)
  __bf16* xb    = (__bf16*)d_ws;                   // [B][D]   8 MB
  __bf16* VbT   = xb + (size_t)B * D;              // [D][H]   8 MB
  __bf16* Vsq   = VbT + (size_t)D * H;             // [D][D]   2 MB
  __bf16* Wkb   = Vsq + (size_t)D * D;             // [H][D]   8 MB (per-l reuse)
  __bf16* WkT   = Wkb + (size_t)H * D;             // [D][H]   8 MB (per-l reuse)
  __bf16* sigma = WkT + (size_t)D * H;             // [B][H]  32 MB (per-l reuse)
  float*  sbuf  = (float*)(sigma + (size_t)B * H); // [B]     16 KB

  const dim3 blk(256);

  // xb = bf16(x)
  mgn_cvt_bf16<<<dim3((B * D / 8) / 256), blk, 0, stream>>>(x, xb, B * D / 8);
  // VbT = bf16(V^T); direct-converted copy lands in Wkb as scratch (unused)
  mgn_cvt_dual<<<dim3(D / 32, H / 32), blk, 0, stream>>>(V, Wkb, VbT, H, D);
  // Vsq = VbT @ VbT^T
  mgn_gemm_vsq<<<dim3(D / 256, D / 128), blk, 0, stream>>>(VbT, Vsq);
  // out = a + xb @ Vsq
  mgn_gemm_out0<<<dim3(D / 256, B / 128), blk, 0, stream>>>(xb, Vsq, a, out);

  for (int l = 0; l < L; l++) {
    const float* Wkl = Wk + (size_t)l * H * D;
    mgn_zero_f32<<<dim3(B / 256), blk, 0, stream>>>(sbuf, B);
    // Wkb = bf16(Wk_l), WkT = bf16(Wk_l^T) in one pass over the f32 weights
    mgn_cvt_dual<<<dim3(D / 32, H / 32), blk, 0, stream>>>(Wkl, Wkb, WkT, H, D);
    mgn_gemm1_act<<<dim3(H / 256, B / 128), blk, 0, stream>>>(xb, Wkb, bk + (size_t)l * H,
                                                              sigma, sbuf);
    mgn_gemm2_acc<<<dim3(D / 256, B / 128), blk, 0, stream>>>(sigma, WkT, sbuf, out);
  }
}
